// CrossAttention_33200097198639
// MI455X (gfx1250) — compile-verified, matrix operand order checked by reading, not run on hardware
//
#include <hip/hip_runtime.h>

typedef _Float16 v8h  __attribute__((ext_vector_type(8)));
typedef _Float16 v16h __attribute__((ext_vector_type(16)));
typedef float    v8f  __attribute__((ext_vector_type(8)));

union F16x16 { v16h v; v8h h[2]; };

constexpr int BB  = 2;
constexpr int LQ  = 2048;
constexpr int LKV = 2048;
constexpr int E   = 1024;
constexpr int H   = 16;
constexpr int D   = 64;
constexpr int M_ROWS = BB * LQ;  // 4096

__device__ __forceinline__ v8f wmma32(v16h a, v16h b, v8f c) {
  return __builtin_amdgcn_wmma_f32_16x16x32_f16(false, a, false, b, (short)0, c,
                                                false, false);
}

// Reductions across the 16-lane N-group (wave32: lanes 0-15 / 16-31 reduce
// independently, matching the C-fragment row layout).
__device__ __forceinline__ float rowmax16(float v) {
  v = fmaxf(v, __shfl_xor(v, 1, 32));
  v = fmaxf(v, __shfl_xor(v, 2, 32));
  v = fmaxf(v, __shfl_xor(v, 4, 32));
  v = fmaxf(v, __shfl_xor(v, 8, 32));
  return v;
}
__device__ __forceinline__ float rowsum16(float v) {
  v += __shfl_xor(v, 1, 32);
  v += __shfl_xor(v, 2, 32);
  v += __shfl_xor(v, 4, 32);
  v += __shfl_xor(v, 8, 32);
  return v;
}

__global__ void __launch_bounds__(256)
cvt_f32_f16(const float* __restrict__ src, _Float16* __restrict__ dst, int n) {
  for (int i = blockIdx.x * blockDim.x + threadIdx.x; i < n;
       i += gridDim.x * blockDim.x)
    dst[i] = (_Float16)src[i];
}

// Y = X @ W^T + b with 32x64 register blocking per wave.
// X:[M_ROWS,E] f16 row-major, W:[E,E] f16 row-major (row n = out column n).
// mode 0: f16 row-major out.  mode 1: f16 V^T out [(b*E+n)*LKV + l].
// mode 2: f32 row-major out (final projection).
__global__ void __launch_bounds__(256)
gemm16(const _Float16* __restrict__ A, const _Float16* __restrict__ W,
       const float* __restrict__ bias, void* __restrict__ dst, int mode) {
  const int wave = threadIdx.x >> 5, lane = threadIdx.x & 31;
  const int col = lane & 15, half = lane >> 4;
  const int tilesN = E / 64;                     // 16
  const int total = (M_ROWS / 32) * tilesN;      // 2048 waves
  const int tid = blockIdx.x * (blockDim.x >> 5) + wave;
  if (tid >= total) return;
  const int tM = tid / tilesN, tN = tid % tilesN;

  const _Float16* arow0 = A + (size_t)(tM * 32 + col) * E;
  const _Float16* arow1 = arow0 + (size_t)16 * E;
  const _Float16* wrow = W + (size_t)(tN * 64 + col) * E;

  v8f acc[2][4] = {};
  for (int k = 0; k < E; k += 32) {
    F16x16 a0, a1, b[4];
    a0.h[0] = *(const v8h*)(arow0 + k + 8 * half);
    a0.h[1] = *(const v8h*)(arow0 + k + 16 + 8 * half);
    a1.h[0] = *(const v8h*)(arow1 + k + 8 * half);
    a1.h[1] = *(const v8h*)(arow1 + k + 16 + 8 * half);
#pragma unroll
    for (int j = 0; j < 4; j++) {
      const _Float16* wr = wrow + (size_t)(j * 16) * E + k;
      b[j].h[0] = *(const v8h*)(wr + 8 * half);
      b[j].h[1] = *(const v8h*)(wr + 16 + 8 * half);
    }
#pragma unroll
    for (int j = 0; j < 4; j++) {
      acc[0][j] = wmma32(a0.v, b[j].v, acc[0][j]);
      acc[1][j] = wmma32(a1.v, b[j].v, acc[1][j]);
    }
  }

  if (mode == 0) {
    _Float16* out = (_Float16*)dst;
#pragma unroll
    for (int i = 0; i < 2; i++)
#pragma unroll
      for (int j = 0; j < 4; j++) {
        const float bval = bias[tN * 64 + j * 16 + col];
#pragma unroll
        for (int r = 0; r < 8; r++) {
          int m = tM * 32 + i * 16 + r + half * 8;
          out[(size_t)m * E + tN * 64 + j * 16 + col] =
              (_Float16)(acc[i][j][r] + bval);
        }
      }
  } else if (mode == 1) {
    _Float16* out = (_Float16*)dst;
#pragma unroll
    for (int i = 0; i < 2; i++)
#pragma unroll
      for (int j = 0; j < 4; j++) {
        const float bval = bias[tN * 64 + j * 16 + col];
#pragma unroll
        for (int r = 0; r < 8; r++) {
          int m = tM * 32 + i * 16 + r + half * 8;
          int b_ = m / LKV, l = m % LKV;
          int n = tN * 64 + j * 16 + col;
          out[((size_t)(b_ * E + n)) * LKV + l] = (_Float16)(acc[i][j][r] + bval);
        }
      }
  } else {
    float* out = (float*)dst;
#pragma unroll
    for (int i = 0; i < 2; i++)
#pragma unroll
      for (int j = 0; j < 4; j++) {
        const float bval = bias[tN * 64 + j * 16 + col];
#pragma unroll
        for (int r = 0; r < 8; r++) {
          int m = tM * 32 + i * 16 + r + half * 8;
          out[(size_t)m * E + tN * 64 + j * 16 + col] = acc[i][j][r] + bval;
        }
      }
  }
}

// Flash-style attention: one wave owns a 32-row Q block of one (b,h).
// Streams KV in chunks of 32; K/V fragments are reused by both 16-row Q
// subtiles. Scores never touch HBM.
__global__ void __launch_bounds__(256)
attn16(const _Float16* __restrict__ Qp,  // [BB*LQ, E]
       const _Float16* __restrict__ Kp,  // [BB*LKV, E]
       const _Float16* __restrict__ Vt,  // [BB*E, LKV]  (V transposed)
       const int* __restrict__ mask,     // [BB, LKV]
       _Float16* __restrict__ ctx)       // [BB*LQ, E]
{
  __shared__ __align__(16) _Float16 plds[8][2][16 * 32];
  const int wave = threadIdx.x >> 5, lane = threadIdx.x & 31;
  const int col = lane & 15, half = lane >> 4;
  const int qBlocks = LQ / 32;                   // 64
  const int id = blockIdx.x * (blockDim.x >> 5) + wave;
  if (id >= BB * H * qBlocks) return;
  const int qb = id % qBlocks;
  const int hb = id / qBlocks;
  const int h = hb % H, b = hb / H;

  // Q block as 2 subtiles x 2 A-fragments (d = 0..31, 32..63).
  F16x16 qa[2][2];
#pragma unroll
  for (int qi = 0; qi < 2; qi++) {
    const _Float16* qrow =
        Qp + ((size_t)(b * LQ + qb * 32 + qi * 16 + col)) * E + h * D;
    qa[qi][0].h[0] = *(const v8h*)(qrow + 8 * half);
    qa[qi][0].h[1] = *(const v8h*)(qrow + 16 + 8 * half);
    qa[qi][1].h[0] = *(const v8h*)(qrow + 32 + 8 * half);
    qa[qi][1].h[1] = *(const v8h*)(qrow + 48 + 8 * half);
  }

  v8f o[2][4] = {};
  float mrow[2][8], lrow[2][8];
#pragma unroll
  for (int qi = 0; qi < 2; qi++)
#pragma unroll
    for (int r = 0; r < 8; r++) { mrow[qi][r] = -1e30f; lrow[qi][r] = 0.0f; }

  const int* mrowp = mask + b * LKV;

  for (int kv = 0; kv < LKV; kv += 32) {
    // K fragments for this kv chunk: 2 column tiles x 2 d-fragments,
    // shared by both Q subtiles.
    F16x16 kb[2][2];
    {
      const _Float16* k0 = Kp + ((size_t)(b * LKV + kv + col)) * E + h * D;
#pragma unroll
      for (int t = 0; t < 2; t++) {
        const _Float16* kr = k0 + (size_t)(t * 16) * E;
        kb[t][0].h[0] = *(const v8h*)(kr + 8 * half);
        kb[t][0].h[1] = *(const v8h*)(kr + 16 + 8 * half);
        kb[t][1].h[0] = *(const v8h*)(kr + 32 + 8 * half);
        kb[t][1].h[1] = *(const v8h*)(kr + 48 + 8 * half);
      }
    }
    const int mk0 = mrowp[kv + col];        // mask depends on kv column only
    const int mk1 = mrowp[kv + 16 + col];

    F16x16 pfrag[2];
#pragma unroll
    for (int qi = 0; qi < 2; qi++) {
      v8f s0 = {}, s1 = {};
      s0 = wmma32(qa[qi][0].v, kb[0][0].v, s0);
      s0 = wmma32(qa[qi][1].v, kb[0][1].v, s0);
      s1 = wmma32(qa[qi][0].v, kb[1][0].v, s1);
      s1 = wmma32(qa[qi][1].v, kb[1][1].v, s1);

      _Float16* lp = plds[wave][qi];
#pragma unroll
      for (int r = 0; r < 8; r++) {
        float sa = mk0 ? s0[r] * 0.125f : -1e30f;   // 1/sqrt(D) = 0.125
        float sb = mk1 ? s1[r] * 0.125f : -1e30f;
        float mn = fmaxf(mrow[qi][r], rowmax16(fmaxf(sa, sb)));
        float corr = __expf(mrow[qi][r] - mn);
        mrow[qi][r] = mn;
        float pa_ = __expf(sa - mn);
        float pb_ = __expf(sb - mn);
        lrow[qi][r] = lrow[qi][r] * corr + rowsum16(pa_ + pb_);
        o[qi][0][r] *= corr; o[qi][1][r] *= corr;
        o[qi][2][r] *= corr; o[qi][3][r] *= corr;
        // C-layout -> LDS row-major tile for relayout into A-fragment
        lp[(r + half * 8) * 32 + col]      = (_Float16)pa_;
        lp[(r + half * 8) * 32 + 16 + col] = (_Float16)pb_;
      }
      pfrag[qi].h[0] = *(const v8h*)(lp + col * 32 + 8 * half);
      pfrag[qi].h[1] = *(const v8h*)(lp + col * 32 + 16 + 8 * half);
    }

    // V^T makes each B-fragment column kv-contiguous; fragments shared by
    // both Q subtiles.
    const _Float16* vb = Vt + ((size_t)(b * E + h * D + col)) * LKV + kv;
#pragma unroll
    for (int dt = 0; dt < 4; dt++) {
      F16x16 vf;
      const _Float16* vr = vb + (size_t)(dt * 16) * LKV;
      vf.h[0] = *(const v8h*)(vr + 8 * half);
      vf.h[1] = *(const v8h*)(vr + 16 + 8 * half);
      o[0][dt] = wmma32(pfrag[0].v, vf.v, o[0][dt]);
      o[1][dt] = wmma32(pfrag[1].v, vf.v, o[1][dt]);
    }
  }

#pragma unroll
  for (int qi = 0; qi < 2; qi++)
#pragma unroll
    for (int r = 0; r < 8; r++) {
      float inv = 1.0f / lrow[qi][r];
      _Float16* cr = ctx +
          ((size_t)(b * LQ + qb * 32 + qi * 16 + r + half * 8)) * E + h * D;
      cr[col]      = (_Float16)(o[qi][0][r] * inv);
      cr[16 + col] = (_Float16)(o[qi][1][r] * inv);
      cr[32 + col] = (_Float16)(o[qi][2][r] * inv);
      cr[48 + col] = (_Float16)(o[qi][3][r] * inv);
    }
}

extern "C" void kernel_launch(void* const* d_in, const int* in_sizes, int n_in,
                              void* d_out, int out_size, void* d_ws,
                              size_t ws_size, hipStream_t stream) {
  (void)in_sizes; (void)n_in; (void)out_size; (void)ws_size;
  const float* query = (const float*)d_in[0];
  const float* keyv  = (const float*)d_in[1];
  const int*   mask  = (const int*)d_in[2];
  const float* Wq = (const float*)d_in[3];
  const float* bq = (const float*)d_in[4];
  const float* Wk = (const float*)d_in[5];
  const float* bk = (const float*)d_in[6];
  const float* Wv = (const float*)d_in[7];
  const float* bv = (const float*)d_in[8];
  const float* Wo = (const float*)d_in[9];
  const float* bo = (const float*)d_in[10];

  const size_t XE = (size_t)M_ROWS * E;  // 4,194,304 halves
  const size_t WE = (size_t)E * E;       // 1,048,576 halves
  _Float16* p = (_Float16*)d_ws;
  _Float16* qf  = p; p += XE;   // query f16
  _Float16* kvf = p; p += XE;   // key_value f16
  _Float16* wqf = p; p += WE;
  _Float16* wkf = p; p += WE;
  _Float16* wvf = p; p += WE;
  _Float16* wof = p; p += WE;
  _Float16* Qp  = p; p += XE;   // Q projection
  _Float16* Kp  = p; p += XE;   // K projection
  _Float16* Vt  = p; p += XE;   // V projection, transposed [B,E,LKV]
  _Float16* ctx = p; p += XE;   // attention output

  cvt_f32_f16<<<2048, 256, 0, stream>>>(query, qf, (int)XE);
  cvt_f32_f16<<<2048, 256, 0, stream>>>(keyv, kvf, (int)XE);
  cvt_f32_f16<<<1024, 256, 0, stream>>>(Wq, wqf, (int)WE);
  cvt_f32_f16<<<1024, 256, 0, stream>>>(Wk, wkf, (int)WE);
  cvt_f32_f16<<<1024, 256, 0, stream>>>(Wv, wvf, (int)WE);
  cvt_f32_f16<<<1024, 256, 0, stream>>>(Wo, wof, (int)WE);

  const int gemmBlocks = ((M_ROWS / 32) * (E / 64) + 7) / 8;  // 256
  gemm16<<<gemmBlocks, 256, 0, stream>>>(qf,  wqf, bq, (void*)Qp, 0);
  gemm16<<<gemmBlocks, 256, 0, stream>>>(kvf, wkf, bk, (void*)Kp, 0);
  gemm16<<<gemmBlocks, 256, 0, stream>>>(kvf, wvf, bv, (void*)Vt, 1);

  attn16<<<(BB * H * (LQ / 32) + 7) / 8, 256, 0, stream>>>(Qp, Kp, Vt, mask,
                                                           ctx);

  gemm16<<<gemmBlocks, 256, 0, stream>>>(ctx, wof, bo, d_out, 2);
}
